// RotatedLinear_90194313216253
// MI455X (gfx1250) — matrix operand order repaired and checked by reading, not compiled, not run on hardware
//
#include <hip/hip_runtime.h>

typedef __attribute__((ext_vector_type(16))) __bf16 v16bf;
typedef __attribute__((ext_vector_type(8)))  __bf16 v8bf;
typedef __attribute__((ext_vector_type(8)))  float  v8f;

// ---------- helpers: fp32 -> bf16 (round-to-nearest-even) ----------
__device__ __forceinline__ unsigned int f2bf_bits(float f) {
    unsigned int u = __float_as_uint(f);
    u += 0x7FFFu + ((u >> 16) & 1u);
    return u >> 16;
}
__device__ __forceinline__ unsigned int pack2bf(float lo, float hi) {
    return f2bf_bits(lo) | (f2bf_bits(hi) << 16);
}

// =====================================================================
// Kernel 1: per-row fast Walsh-Hadamard transform (n = 8192, fp32 in LDS)
// then scale by 1/sqrt(n) and store the row as bf16 (row-major).
// One workgroup (256 threads, 8 waves) per row; row = 32KB in LDS.
// =====================================================================
#define HN   8192
#define HT   256
#define HPER 32     // HN / HT elements per thread

__global__ __launch_bounds__(HT) void fwht_rows_to_bf16(
        const float* __restrict__ x, __bf16* __restrict__ xh) {
    __shared__ float lds[HN];
    const int row = blockIdx.x;
    const int t   = threadIdx.x;

    float v[HPER];
    const float4* src = (const float4*)(x + (size_t)row * HN + t * HPER);
    #pragma unroll
    for (int q = 0; q < HPER / 4; ++q) {
        float4 f = src[q];
        v[q*4+0] = f.x; v[q*4+1] = f.y; v[q*4+2] = f.z; v[q*4+3] = f.w;
    }

    // Stages h = 1,2,4,8,16 fully in registers (contiguous ownership).
    #pragma unroll
    for (int h = 1; h < HPER; h <<= 1) {
        #pragma unroll
        for (int i = 0; i < HPER; i += 2 * h) {
            #pragma unroll
            for (int j = i; j < i + h; ++j) {
                float a = v[j], b = v[j + h];
                v[j] = a + b; v[j + h] = a - b;
            }
        }
    }
    #pragma unroll
    for (int j = 0; j < HPER; ++j) lds[t * HPER + j] = v[j];

    // Stages h = 32 .. 4096 via LDS (bank-conflict free).
    for (int h = HPER; h < HN; h <<= 1) {
        __syncthreads();
        #pragma unroll 4
        for (int p = t; p < HN / 2; p += HT) {
            int i = ((p & ~(h - 1)) << 1) | (p & (h - 1));
            float a = lds[i], b = lds[i + h];
            lds[i] = a + b; lds[i + h] = a - b;
        }
    }
    __syncthreads();

    const float s = 0.011048543456f;  // 1/sqrt(8192)
    unsigned int o[HPER / 2];
    #pragma unroll
    for (int j = 0; j < HPER / 2; ++j)
        o[j] = pack2bf(lds[t * HPER + 2 * j] * s, lds[t * HPER + 2 * j + 1] * s);
    uint4* dst = (uint4*)(xh + (size_t)row * HN + t * HPER);
    #pragma unroll
    for (int q = 0; q < HPER / 8; ++q)
        dst[q] = make_uint4(o[q*4+0], o[q*4+1], o[q*4+2], o[q*4+3]);
}

// =====================================================================
// Kernel 2: W fp32 -> bf16 (8 elements per thread, vectorized)
// =====================================================================
__global__ __launch_bounds__(256) void w_to_bf16(
        const float* __restrict__ W, __bf16* __restrict__ Wb, long n) {
    long idx = ((long)blockIdx.x * 256 + threadIdx.x) * 8;
    if (idx >= n) return;
    const float4* src = (const float4*)(W + idx);
    float4 f0 = src[0], f1 = src[1];
    uint4 o;
    o.x = pack2bf(f0.x, f0.y); o.y = pack2bf(f0.z, f0.w);
    o.z = pack2bf(f1.x, f1.y); o.w = pack2bf(f1.z, f1.w);
    *(uint4*)(Wb + idx) = o;
}

// =====================================================================
// Kernel 3: out[M,N] = A[M,K](bf16) * B[N,K](bf16)^T + bias, fp32 accum.
// Wave tile 64x64 = 4x4 WMMA tiles -> 16 v_wmma per K-step of 32 against
// 8 fragment loads (32 FLOP / fetched byte). K is constexpr so all
// fragment addresses fold into one base + 24-bit immediate offsets.
// 8 waves arranged 2(M) x 4(N) -> 128x256 per workgroup.
// =====================================================================
template <int K>
__global__ __launch_bounds__(256) void gemm_xhW(
        const __bf16* __restrict__ A,    // [M,K]
        const __bf16* __restrict__ B,    // [N,K] (rows of W)
        const float*  __restrict__ bias, // [N]
        float*        __restrict__ out,  // [M,N]
        int N) {
    const int lane = threadIdx.x & 31;
    const int wave = threadIdx.x >> 5;   // 0..7
    const int l16  = lane & 15;
    const int half = lane >> 4;          // 0 or 1

    const int tileM = blockIdx.y * 128 + (wave >> 2) * 64;   // 2 waves in M
    const int tileN = blockIdx.x * 256 + (wave & 3) * 64;    // 4 waves in N

    // Per-lane fragment bases (A and B both use contiguous k = half*16 + s):
    //  A lane: row tileM+l16 (+16*i), chunks at k+half*16 and +8.
    //  B lane: W-row tileN+l16 (+16*j), 32 contiguous bytes at k+half*16.
    const __bf16* aBase = A + (size_t)(tileM + l16) * K + half * 16;
    const __bf16* bBase = B + (size_t)(tileN + l16) * K + half * 16;

    v8f c[4][4];
    #pragma unroll
    for (int i = 0; i < 4; ++i)
        #pragma unroll
        for (int j = 0; j < 4; ++j) c[i][j] = (v8f){};

    #pragma unroll 1
    for (int k = 0; k < K; k += 32) {
        v16bf fa[4], fb[4];
        #pragma unroll
        for (int i = 0; i < 4; ++i) {
            // constant row offset i*16*K folds into the 24-bit ioffset
            v8bf lo = *(const v8bf*)(aBase + (size_t)i * 16 * K + k);
            v8bf hi = *(const v8bf*)(aBase + (size_t)i * 16 * K + k + 8);
            fa[i] = __builtin_shufflevector(lo, hi,
                        0,1,2,3,4,5,6,7,8,9,10,11,12,13,14,15);
        }
        #pragma unroll
        for (int j = 0; j < 4; ++j)
            fb[j] = *(const v16bf*)(bBase + (size_t)j * 16 * K + k);

        #pragma unroll
        for (int i = 0; i < 4; ++i)
            #pragma unroll
            for (int j = 0; j < 4; ++j)
                c[i][j] = __builtin_amdgcn_wmma_f32_16x16x32_bf16(
                              false, fa[i], false, fb[j],
                              (short)0, c[i][j], false, false);
    }

    // C/D layout: VGPR r, lanes 0-15: (M=tile+r, N=tile+lane);
    // lanes 16-31: (M=tile+8+r, N=tile+lane-16).
    #pragma unroll
    for (int j = 0; j < 4; ++j) {
        const int n  = tileN + 16 * j + l16;
        const float bv = bias[n];
        #pragma unroll
        for (int i = 0; i < 4; ++i) {
            const int m0 = tileM + 16 * i + half * 8;
            #pragma unroll
            for (int r = 0; r < 8; ++r)
                out[(size_t)(m0 + r) * N + n] = c[i][j][r] + bv;
        }
    }
}

// =====================================================================
extern "C" void kernel_launch(void* const* d_in, const int* in_sizes, int n_in,
                              void* d_out, int out_size, void* d_ws, size_t ws_size,
                              hipStream_t stream) {
    const float* x = (const float*)d_in[0];
    const float* W = (const float*)d_in[1];
    const float* b = (const float*)d_in[2];
    float* out = (float*)d_out;

    constexpr int K = 8192;                     // D_IN (fixed by problem)
    const int  N = in_sizes[2];                 // 2048
    const long M = (long)in_sizes[0] / K;       // 8192 (= B*S)

    __bf16* xh = (__bf16*)d_ws;                                 // M*K bf16
    __bf16* Wb = (__bf16*)((char*)d_ws + (size_t)M * K * 2);    // N*K bf16

    // 1) Hadamard rotate rows (fp32) -> bf16 staging
    fwht_rows_to_bf16<<<(int)M, HT, 0, stream>>>(x, xh);

    // 2) W -> bf16
    long wn = (long)in_sizes[1];
    int wblocks = (int)((wn / 8 + 255) / 256);
    w_to_bf16<<<wblocks, 256, 0, stream>>>(W, Wb, wn);

    // 3) WMMA GEMM + bias: grid over (N/256, M/128)
    dim3 grid((unsigned)(N / 256), (unsigned)(M / 128));
    gemm_xhW<K><<<grid, 256, 0, stream>>>(xh, Wb, b, out, N);
}
